// ARTransformer_76759655514669
// MI455X (gfx1250) — matrix operand order repaired and checked by reading, not compiled, not run on hardware
//
#include <hip/hip_runtime.h>

// ---------------------------------------------------------------------------
// Types for CDNA5 WMMA (wave32)
// ---------------------------------------------------------------------------
typedef __bf16 bf16;
typedef __attribute__((ext_vector_type(16))) __bf16 v16bf;
typedef __attribute__((ext_vector_type(8)))  __bf16 bf16x8;
typedef __attribute__((ext_vector_type(4)))  __bf16 bf16x4;
typedef __attribute__((ext_vector_type(8)))  float  v8f;
typedef unsigned int u32x4 __attribute__((ext_vector_type(4)));
typedef int i32x4v __attribute__((ext_vector_type(4)));
typedef int i32x8v __attribute__((ext_vector_type(8)));

union FragU { v16bf v; bf16x8 h[2]; };

#define B_   8
#define LS_  512
#define L_   1024
#define D_   1024
#define H_   16
#define HD_  64
#define FF_  4096
#define NL_  12
#define MTOT (B_ * L_)   // 8192 token rows

#if __has_builtin(__builtin_amdgcn_tensor_load_to_lds) && \
    __has_builtin(__builtin_amdgcn_s_wait_tensorcnt)
#define HAVE_TDM 1
#else
#define HAVE_TDM 0
#endif

__device__ __forceinline__ v8f wmma_bf16(v16bf a, v16bf b, v8f c) {
  // D = A(16x32 bf16) * B(32x16 bf16) + C(16x16 f32)
  return __builtin_amdgcn_wmma_f32_16x16x32_bf16(false, a, false, b,
                                                 (short)0, c, false, false);
}

__device__ __forceinline__ int rfl(int v) {
  return __builtin_amdgcn_readfirstlane(v);
}

#if HAVE_TDM
// LDS byte offset of a generic pointer to a __shared__ object
__device__ __forceinline__ unsigned lds_off(const void* p) {
  return (unsigned)(unsigned long long)
      (__attribute__((address_space(3))) const void*)p;
}

// Tensor Data Mover: async copy of a 2-D bf16 tile (tile_cols x tile_rows)
// from row-major global (leading dim ld_elems) into contiguous LDS.
// D# packing per CDNA5 ISA.
__device__ __forceinline__ void tdm_load_tile_2d(unsigned ldsoff,
                                                 const bf16* gsrc,
                                                 int tile_rows, int tile_cols,
                                                 int ld_elems, int tensor_rows) {
  unsigned long long ga = (unsigned long long)gsrc;
  u32x4 g0;
  g0[0] = 1u;                                    // count=1 (valid), user mode
  g0[1] = ldsoff;                                // lds_addr (bytes)
  g0[2] = (unsigned)ga;                          // global_addr[31:0]
  g0[3] = (unsigned)((ga >> 32) & 0x01ffffffu)   // global_addr[56:32]
        | (2u << 30);                            // type = 2 ("image")
  unsigned td0 = (unsigned)ld_elems;             // tensor_dim0 (elements)
  unsigned td1 = (unsigned)tensor_rows;          // tensor_dim1
  i32x8v g1;
  g1[0] = (int)(1u << 16);                       // data_size=1 (2 bytes)
  g1[1] = (int)((td0 & 0xffffu) << 16);          // tensor_dim0 lo16
  g1[2] = (int)((td0 >> 16) | ((td1 & 0xffffu) << 16));
  g1[3] = (int)((td1 >> 16) | ((unsigned)tile_cols << 16));  // tile_dim0
  g1[4] = (int)(unsigned)tile_rows;              // tile_dim1 (tile_dim2 = 0)
  g1[5] = (int)td0;                              // tensor_dim0_stride lo32
  g1[6] = 0;                                     // stride hi / dim1_stride lo
  g1[7] = 0;
  i32x4v z4 = {0, 0, 0, 0};
#if defined(__clang_major__) && __clang_major__ >= 23
  i32x8v z8 = {0, 0, 0, 0, 0, 0, 0, 0};
  __builtin_amdgcn_tensor_load_to_lds(g0, g1, z4, z4, z8, 0);
#else
  __builtin_amdgcn_tensor_load_to_lds(g0, g1, z4, z4, 0);
#endif
}
#endif  // HAVE_TDM

// ---------------------------------------------------------------------------
// x[b,l,:] = concat(sensor, traj)[b,l,:] + pos_embed[l,:]
// ---------------------------------------------------------------------------
__global__ void k_embed(const float* __restrict__ sensor,
                        const float* __restrict__ traj,
                        const float* __restrict__ pos,
                        float* __restrict__ x) {
  int row = blockIdx.x;                 // 0..8191
  int b = row >> 10, l = row & 1023;
  const float* src = (l < LS_) ? (sensor + ((size_t)b * LS_ + l) * D_)
                               : (traj   + ((size_t)b * LS_ + (l - LS_)) * D_);
  const float* pp = pos + (size_t)l * D_;
  float* dst = x + (size_t)row * D_;
  for (int d = threadIdx.x; d < D_; d += blockDim.x) dst[d] = src[d] + pp[d];
}

// ---------------------------------------------------------------------------
// fp32 -> bf16 weight conversion (per-layer, into rotating scratch)
// ---------------------------------------------------------------------------
__global__ void k_cvt(const float* __restrict__ s, bf16* __restrict__ d, int n) {
  int i = blockIdx.x * blockDim.x + threadIdx.x;
  int stride = gridDim.x * blockDim.x;
  for (; i < n; i += stride) d[i] = (bf16)s[i];
}

// ---------------------------------------------------------------------------
// LayerNorm: one 256-thread block per token row (D=1024, float4 per thread)
// ---------------------------------------------------------------------------
__global__ __launch_bounds__(256)
void k_ln(const float* __restrict__ x, const float* __restrict__ sc,
          const float* __restrict__ bi, bf16* __restrict__ obf,
          float* __restrict__ of32, int f32out) {
  int row = blockIdx.x;
  const float4* xr = (const float4*)(x + (size_t)row * D_);
  float4 v = xr[threadIdx.x];
  float s  = v.x + v.y + v.z + v.w;
  float s2 = v.x * v.x + v.y * v.y + v.z * v.z + v.w * v.w;
#pragma unroll
  for (int o = 16; o; o >>= 1) {
    s  += __shfl_down(s,  o, 32);
    s2 += __shfl_down(s2, o, 32);
  }
  __shared__ float rs[8], rq[8];
  int wid = threadIdx.x >> 5, lane = threadIdx.x & 31;
  if (lane == 0) { rs[wid] = s; rq[wid] = s2; }
  __syncthreads();
  if (threadIdx.x == 0) {
    float a = 0.f, b2 = 0.f;
#pragma unroll
    for (int i = 0; i < 8; i++) { a += rs[i]; b2 += rq[i]; }
    rs[0] = a; rq[0] = b2;
  }
  __syncthreads();
  float mean = rs[0] * (1.0f / D_);
  float var  = rq[0] * (1.0f / D_) - mean * mean;
  float r    = rsqrtf(var + 1e-5f);
  float4 c4 = ((const float4*)sc)[threadIdx.x];
  float4 b4 = ((const float4*)bi)[threadIdx.x];
  float o0 = (v.x - mean) * r * c4.x + b4.x;
  float o1 = (v.y - mean) * r * c4.y + b4.y;
  float o2 = (v.z - mean) * r * c4.z + b4.z;
  float o3 = (v.w - mean) * r * c4.w + b4.w;
  if (f32out) {
    float4 o4 = {o0, o1, o2, o3};
    ((float4*)(of32 + (size_t)row * D_))[threadIdx.x] = o4;
  } else {
    bf16x4 ob; ob[0] = (bf16)o0; ob[1] = (bf16)o1; ob[2] = (bf16)o2; ob[3] = (bf16)o3;
    ((bf16x4*)(obf + (size_t)row * D_))[threadIdx.x] = ob;
  }
}

// ---------------------------------------------------------------------------
// GEMM: C[M,N] = A[M,K](bf16,rm) * W[N,K](bf16,rm)^T + bias
// 8 waves/block; wave tile 32x64 (2x4 C-tiles); block tile 64x256.
// The 256x32 B-tile is staged into LDS by the Tensor Data Mover, double
// buffered: wave 0 issues the next tile's DMA while all waves compute.
// EPI 0: QKV scatter (Q scaled 1/8, V transposed), EPI 1: GELU->bf16,
// EPI 2: residual add into fp32 x.
// ---------------------------------------------------------------------------
template <int EPI>
__global__ __launch_bounds__(256)
void k_gemm(const bf16* __restrict__ A, int lda,
            const bf16* __restrict__ W, int ldw,
            const float* __restrict__ bias, int K, int Ntot,
            float* __restrict__ xres, bf16* __restrict__ obf, int ldo,
            bf16* __restrict__ qb, bf16* __restrict__ kb, bf16* __restrict__ vt) {
  __shared__ alignas(16) bf16 smB[2][256 * 32];   // 2 x 16 KB
  const int tid  = threadIdx.x;
  const int wave = tid >> 5, lane = tid & 31;
  const int r = lane & 15, hf = lane >> 4;
  const int mbase  = rfl(blockIdx.x * 64 + (wave & 1) * 32);
  const int nbase0 = rfl(blockIdx.y * 256);       // block's N base
  const int nloc   = rfl((wave >> 1) * 64);       // wave's N offset in block

  v8f acc[2][4];
#pragma unroll
  for (int i = 0; i < 2; i++)
#pragma unroll
    for (int j = 0; j < 4; j++) acc[i][j] = (v8f){0.f,0.f,0.f,0.f,0.f,0.f,0.f,0.f};

#if HAVE_TDM
  if (wave == 0)
    tdm_load_tile_2d(lds_off(&smB[0][0]), W + (size_t)nbase0 * ldw,
                     256, 32, ldw, Ntot - nbase0);
#endif

  for (int k0 = 0; k0 < K; k0 += 32) {
#if HAVE_TDM
    const int cur = (k0 >> 5) & 1;
    if (wave == 0) __builtin_amdgcn_s_wait_tensorcnt(0);
    __syncthreads();   // buf[cur] ready; everyone done reading buf[cur^1]
    if (wave == 0 && (k0 + 32) < K)
      tdm_load_tile_2d(lds_off(&smB[cur ^ 1][0]),
                       W + (size_t)nbase0 * ldw + (k0 + 32),
                       256, 32, ldw, Ntot - nbase0);
#else
    const int cur = 0;
    __syncthreads();   // protect previous iteration's reads
    {  // cooperative staging: thread t copies k 0..7, 16..23 of row t (16 bf16 = 32B)
      const bf16* src = W + (size_t)(nbase0 + tid) * ldw + k0;
      *(uint4*)&smB[0][tid * 32]      = *(const uint4*)src;
      *(uint4*)&smB[0][tid * 32 + 16] = *(const uint4*)(src + 16);
    }
    __syncthreads();
#endif
    if (k0 + 32 < K) {   // cache hint for next A tile (global_prefetch_b8)
      __builtin_prefetch(A + (size_t)(mbase + r) * lda + k0 + 32, 0, 1);
    }
    FragU a[2];
#pragma unroll
    for (int i = 0; i < 2; i++) {
      const bf16* p = A + (size_t)(mbase + 16 * i + r) * lda + k0 + hf * 8;
      a[i].h[0] = *(const bf16x8*)p;          // K = h*8 .. h*8+7
      a[i].h[1] = *(const bf16x8*)(p + 16);   // K = 16+h*8 ..
    }
    FragU b[4];
#pragma unroll
    for (int j = 0; j < 4; j++) {
      const bf16* p = &smB[cur][(nloc + 16 * j + r) * 32 + hf * 8];
      b[j].h[0] = *(const bf16x8*)p;          // K = h*8 .. h*8+7, 16+h*8 ..
      b[j].h[1] = *(const bf16x8*)(p + 16);
    }
#pragma unroll
    for (int i = 0; i < 2; i++)
#pragma unroll
      for (int j = 0; j < 4; j++)
        acc[i][j] = wmma_bf16(a[i].v, b[j].v, acc[i][j]);
  }

#pragma unroll
  for (int i = 0; i < 2; i++)
#pragma unroll
    for (int j = 0; j < 4; j++) {
      const int ncb = nbase0 + nloc + 16 * j;   // scalar column base of tile
      const int n = ncb + r;
      float bv = bias[n];
#pragma unroll
      for (int v = 0; v < 8; v++) {
        int m = mbase + 16 * i + 8 * hf + v;
        float val = acc[i][j][v] + bv;
        if (EPI == 0) {                       // QKV scatter (uniform selectors)
          const int which = ncb >> 10;        // 0=q 1=k 2=v  (scalar)
          const int hh = (ncb >> 6) & 15;     // head          (scalar)
          const int d = (ncb & 63) + r;       // head dim      (lane)
          int bI = m >> 10, l = m & 1023;
          int bh = bI * H_ + hh;
          if (which == 0)
            qb[((size_t)bh * L_ + l) * HD_ + d] = (bf16)(val * 0.125f); // 1/sqrt(64)
          else if (which == 1)
            kb[((size_t)bh * L_ + l) * HD_ + d] = (bf16)val;
          else
            vt[((size_t)bh * HD_ + d) * L_ + l] = (bf16)val;            // transposed
        } else if (EPI == 1) {                // exact GELU -> bf16
          float g = 0.5f * val * (1.0f + erff(val * 0.70710678118654752f));
          obf[(size_t)m * ldo + n] = (bf16)g;
        } else {                              // residual add (fp32 stream)
          xres[(size_t)m * D_ + n] += val;
        }
      }
    }
}

// ---------------------------------------------------------------------------
// Flash-style attention, transposed formulation (per-lane softmax stats).
// One wave per (b, h, 16-query tile); 8 WMMAs per 32-key tile.
// Prefix-causal mask applied branchlessly per lane:
//   allowed(q, k) = (k <= q) || (q < LS && k < LS)
// ---------------------------------------------------------------------------
__global__ __launch_bounds__(256)
void k_attn(const bf16* __restrict__ Q, const bf16* __restrict__ Kb,
            const bf16* __restrict__ Vt, bf16* __restrict__ attn) {
  int wave = threadIdx.x >> 5, lane = threadIdx.x & 31;
  int task = rfl(blockIdx.x * 8 + wave);       // 0..8191 (wave-uniform scalar)
  int qt = task & 63, hh = (task >> 6) & 15, b = task >> 10;
  int bh = b * H_ + hh;
  int q_base = qt * 16;
  int r = lane & 15, hf = lane >> 4;
  int q = q_base + r;                          // this lane's query
  const bool qSen = q < LS_;                   // sensor query (bidirectional)

  // Q as B-fragments (Kdim = d): lane holds Q[q][8*hf .. 8*hf+7], [16+8*hf ..] (+32)
  const bf16* Qrow = Q + ((size_t)bh * L_ + q) * HD_;
  FragU bq0, bq1;
  bq0.h[0] = *(const bf16x8*)(Qrow + hf * 8);
  bq0.h[1] = *(const bf16x8*)(Qrow + hf * 8 + 16);
  bq1.h[0] = *(const bf16x8*)(Qrow + 32 + hf * 8);
  bq1.h[1] = *(const bf16x8*)(Qrow + 32 + hf * 8 + 16);

  v8f o[4];
#pragma unroll
  for (int t = 0; t < 4; t++) o[t] = (v8f){0.f,0.f,0.f,0.f,0.f,0.f,0.f,0.f};
  float m_run = -1e30f, l_run = 0.f;

  // Keys to visit: sensor queries see the sensor block; traj queries see k<=q.
  const int ntiles = rfl((q_base >= LS_) ? ((q_base + 47) >> 5) : (LS_ / 32));

  for (int t = 0; t < ntiles; t++) {
    int kbp = t * 32;
    // S^T tiles: keys kbp..+15 (lo) and kbp+16..+31 (hi), K dim = d (2x32)
    v8f stlo = (v8f){0.f,0.f,0.f,0.f,0.f,0.f,0.f,0.f};
    v8f sthi = (v8f){0.f,0.f,0.f,0.f,0.f,0.f,0.f,0.f};
    {
      const bf16* Krow = Kb + ((size_t)bh * L_ + kbp + r) * HD_;
      FragU ak;
      ak.h[0] = *(const bf16x8*)(Krow + hf * 8);
      ak.h[1] = *(const bf16x8*)(Krow + hf * 8 + 16);
      stlo = wmma_bf16(ak.v, bq0.v, stlo);
      ak.h[0] = *(const bf16x8*)(Krow + 32 + hf * 8);
      ak.h[1] = *(const bf16x8*)(Krow + 32 + hf * 8 + 16);
      stlo = wmma_bf16(ak.v, bq1.v, stlo);
      Krow += (size_t)16 * HD_;
      ak.h[0] = *(const bf16x8*)(Krow + hf * 8);
      ak.h[1] = *(const bf16x8*)(Krow + hf * 8 + 16);
      sthi = wmma_bf16(ak.v, bq0.v, sthi);
      ak.h[0] = *(const bf16x8*)(Krow + 32 + hf * 8);
      ak.h[1] = *(const bf16x8*)(Krow + 32 + hf * 8 + 16);
      sthi = wmma_bf16(ak.v, bq1.v, sthi);
    }
    // Per-lane: 16 score values for query q; branchless prefix-causal mask
    float slo[8], shi[8];
#pragma unroll
    for (int j = 0; j < 8; j++) {
      int klo = kbp + 8 * hf + j;
      int khi = klo + 16;
      bool alo = (klo <= q) || (qSen && (klo < LS_));
      bool ahi = (khi <= q) || (qSen && (khi < LS_));
      slo[j] = alo ? (float)stlo[j] : -1e30f;
      shi[j] = ahi ? (float)sthi[j] : -1e30f;
    }
    float mt = -1e30f;
#pragma unroll
    for (int j = 0; j < 8; j++) mt = fmaxf(mt, fmaxf(slo[j], shi[j]));
    mt = fmaxf(mt, __shfl_xor(mt, 16, 32));       // merge lane pair (same q)
    float m_new = fmaxf(m_run, mt);
    float corr = __expf(m_run - m_new);
    float plo[8], phi[8];
    float ls = 0.f;
#pragma unroll
    for (int j = 0; j < 8; j++) {
      plo[j] = __expf(slo[j] - m_new);
      phi[j] = __expf(shi[j] - m_new);
      ls += plo[j] + phi[j];
    }
    ls += __shfl_xor(ls, 16, 32);
    l_run = l_run * corr + ls;
    m_run = m_new;
#pragma unroll
    for (int dt = 0; dt < 4; dt++)
#pragma unroll
      for (int v = 0; v < 8; v++) o[dt][v] *= corr;

    // Build P^T B-fragment (32 keys x 16 q). Lane half 0 needs keys 0..15,
    // half 1 needs keys 16..31 -> exchange with partner lane (xor 16).
    float tlo[8], thi[8];
#pragma unroll
    for (int j = 0; j < 8; j++) {
      tlo[j] = __shfl_xor(plo[j], 16, 32);
      thi[j] = __shfl_xor(phi[j], 16, 32);
    }
    FragU bp;
#pragma unroll
    for (int j = 0; j < 8; j++) {
      bp.v[j]     = (bf16)((hf == 0) ? plo[j] : thi[j]);
      bp.v[8 + j] = (bf16)((hf == 0) ? tlo[j] : phi[j]);
    }

    // O^T[d, q] += V^T[d, 32k] * P^T[32k, q]  (V stored transposed: Vt[bh,d,L])
#pragma unroll
    for (int dt = 0; dt < 4; dt++) {
      const bf16* Vrow = Vt + ((size_t)bh * HD_ + dt * 16 + r) * L_ + kbp + hf * 16;
      FragU av;
      av.h[0] = *(const bf16x8*)Vrow;
      av.h[1] = *(const bf16x8*)(Vrow + 8);
      o[dt] = wmma_bf16(av.v, bp.v, o[dt]);
    }
  }

  float inv = 1.0f / l_run;
#pragma unroll
  for (int dt = 0; dt < 4; dt++) {
    int d0 = dt * 16 + 8 * hf;
    bf16x8 ov;
#pragma unroll
    for (int j = 0; j < 8; j++) ov[j] = (bf16)(o[dt][j] * inv);
    *(bf16x8*)(attn + ((size_t)(b * L_ + q_base + r)) * D_ + hh * HD_ + d0) = ov;
  }
}

// ---------------------------------------------------------------------------
// Host orchestration
// ---------------------------------------------------------------------------
extern "C" void kernel_launch(void* const* d_in, const int* in_sizes, int n_in,
                              void* d_out, int out_size, void* d_ws, size_t ws_size,
                              hipStream_t stream) {
  const float* sensor = (const float*)d_in[0];
  const float* traj   = (const float*)d_in[1];
  const float* pos    = (const float*)d_in[2];
  const float* ln1_s  = (const float*)d_in[3];
  const float* ln1_b  = (const float*)d_in[4];
  const float* qkv_w  = (const float*)d_in[5];
  const float* qkv_b  = (const float*)d_in[6];
  const float* out_w  = (const float*)d_in[7];
  const float* out_b  = (const float*)d_in[8];
  const float* ln2_s  = (const float*)d_in[9];
  const float* ln2_b  = (const float*)d_in[10];
  const float* w1     = (const float*)d_in[11];
  const float* b1     = (const float*)d_in[12];
  const float* w2     = (const float*)d_in[13];
  const float* b2     = (const float*)d_in[14];
  const float* fln_s  = (const float*)d_in[15];
  const float* fln_b  = (const float*)d_in[16];

  // Workspace layout (~185 MB)
  size_t off = 0;
  auto walloc = [&](size_t bytes) {
    void* p = (char*)d_ws + off;
    off += (bytes + 255) & ~(size_t)255;
    return p;
  };
  float* x   = (float*)walloc((size_t)MTOT * D_ * 4);    // fp32 residual stream
  bf16* h    = (bf16*) walloc((size_t)MTOT * D_ * 2);    // LN output
  bf16* qb   = (bf16*) walloc((size_t)B_ * H_ * L_ * HD_ * 2);
  bf16* kb   = (bf16*) walloc((size_t)B_ * H_ * L_ * HD_ * 2);
  bf16* vt   = (bf16*) walloc((size_t)B_ * H_ * HD_ * L_ * 2);  // V transposed
  bf16* attn = (bf16*) walloc((size_t)MTOT * D_ * 2);
  bf16* ff1  = (bf16*) walloc((size_t)MTOT * FF_ * 2);
  bf16* wbuf = (bf16*) walloc((size_t)FF_ * D_ * 2);     // rotating bf16 weights
  (void)ws_size; (void)in_sizes; (void)n_in; (void)out_size;

  dim3 blk(256);
  k_embed<<<dim3(MTOT), blk, 0, stream>>>(sensor, traj, pos, x);

  dim3 gQKV(MTOT / 64, (3 * D_) / 256);
  dim3 gOUT(MTOT / 64, D_ / 256);
  dim3 gFF1(MTOT / 64, FF_ / 256);
  dim3 gFF2(MTOT / 64, D_ / 256);

  for (int i = 0; i < NL_; i++) {
    // --- attention (pre-norm) ---
    k_ln<<<dim3(MTOT), blk, 0, stream>>>(x, ln1_s + (size_t)i * D_,
                                         ln1_b + (size_t)i * D_, h, nullptr, 0);
    int nqkv = 3 * D_ * D_;
    k_cvt<<<dim3((nqkv + 255) / 256), blk, 0, stream>>>(
        qkv_w + (size_t)i * 3 * D_ * D_, wbuf, nqkv);
    k_gemm<0><<<gQKV, blk, 0, stream>>>(h, D_, wbuf, D_,
                                        qkv_b + (size_t)i * 3 * D_, D_, 3 * D_,
                                        nullptr, nullptr, 0, qb, kb, vt);
    k_attn<<<dim3((B_ * H_ * (L_ / 16)) / 8), blk, 0, stream>>>(qb, kb, vt, attn);
    int nout = D_ * D_;
    k_cvt<<<dim3((nout + 255) / 256), blk, 0, stream>>>(
        out_w + (size_t)i * D_ * D_, wbuf, nout);
    k_gemm<2><<<gOUT, blk, 0, stream>>>(attn, D_, wbuf, D_,
                                        out_b + (size_t)i * D_, D_, D_,
                                        x, nullptr, 0, nullptr, nullptr, nullptr);
    // --- FFN (pre-norm) ---
    k_ln<<<dim3(MTOT), blk, 0, stream>>>(x, ln2_s + (size_t)i * D_,
                                         ln2_b + (size_t)i * D_, h, nullptr, 0);
    int nw1 = FF_ * D_;
    k_cvt<<<dim3((nw1 + 255) / 256), blk, 0, stream>>>(
        w1 + (size_t)i * FF_ * D_, wbuf, nw1);
    k_gemm<1><<<gFF1, blk, 0, stream>>>(h, D_, wbuf, D_,
                                        b1 + (size_t)i * FF_, D_, FF_,
                                        nullptr, ff1, FF_, nullptr, nullptr, nullptr);
    int nw2 = D_ * FF_;
    k_cvt<<<dim3((nw2 + 255) / 256), blk, 0, stream>>>(
        w2 + (size_t)i * D_ * FF_, wbuf, nw2);
    k_gemm<2><<<gFF2, blk, 0, stream>>>(ff1, FF_, wbuf, FF_,
                                        b2 + (size_t)i * D_, FF_, D_,
                                        x, nullptr, 0, nullptr, nullptr, nullptr);
  }
  // final LN -> fp32 output
  k_ln<<<dim3(MTOT), blk, 0, stream>>>(x, fln_s, fln_b, nullptr, (float*)d_out, 1);
}
